// MixtureOfExperts_49443663512139
// MI455X (gfx1250) — compile-verified, matrix operand order checked by reading, not run on hardware
//
#include <hip/hip_runtime.h>
#include <hip/hip_bf16.h>
#include <math.h>

// Problem constants (reference: B=4, S=2048, D=1024, E=8, F=4096, TOP_K=2)
#define NT 8192   // tokens
#define ND 1024
#define NE 8
#define NF 4096

typedef __attribute__((ext_vector_type(16))) __bf16 v16bf;
typedef __attribute__((ext_vector_type(2)))  __bf16 v2bf;
typedef __attribute__((ext_vector_type(8)))  float  v8f;

__device__ __forceinline__ unsigned short f2bf(float f) {          // exact RNE
    unsigned int u = __float_as_uint(f);
    unsigned int r = u + 0x7FFFu + ((u >> 16) & 1u);
    return (unsigned short)(r >> 16);
}

// Packed f32->bf16: hw cvt_pk if present, else round + v_perm_b32 byte pack (3 VALU).
__device__ __forceinline__ unsigned cvtpk(float a, float b) {
#if __has_builtin(__builtin_amdgcn_cvt_pk_bf16_f32)
    union { v2bf v; unsigned u; } c;
    c.v = __builtin_amdgcn_cvt_pk_bf16_f32(a, b);
    return c.u;
#elif __has_builtin(__builtin_amdgcn_perm)
    unsigned ua = __float_as_uint(a) + 0x8000u;   // round-half-up to bf16
    unsigned ub = __float_as_uint(b) + 0x8000u;
    return __builtin_amdgcn_perm(ub, ua, 0x07060302u); // {ub.hi16, ua.hi16}
#else
    return (unsigned)f2bf(a) | ((unsigned)f2bf(b) << 16);
#endif
}

// K start for dword j of a 16-bit WMMA A/B fragment (ISA 7.12.2 layout).
__device__ __forceinline__ int kstart_of(int j, int half) {
    return (j < 4) ? (8 * half + 2 * j) : (16 + 8 * half + 2 * (j - 4));
}

__device__ __forceinline__ void wait_async0() {
#if __has_builtin(__builtin_amdgcn_s_wait_asynccnt)
    __builtin_amdgcn_s_wait_asynccnt(0);
#else
    asm volatile("s_wait_asynccnt 0" ::: "memory");
#endif
}
__device__ __forceinline__ void async_b128(unsigned lds_off, const void* gaddr) {
    asm volatile("global_load_async_to_lds_b128 %0, %1, off"
                 :: "v"(lds_off), "v"((unsigned long long)(size_t)gaddr) : "memory");
}

// ------------------------------------------------- weight transpose+convert -
// src: [E][M][N] fp32  ->  dst: [E][N][M] bf16   (one-time, memory-bound)
__global__ void moe_tcvt_kernel(const float* __restrict__ src,
                                unsigned short* __restrict__ dst, int M, int N) {
    __shared__ float tile[32][33];
    const int e = blockIdx.z;
    const float* S = src + (size_t)e * M * N;
    unsigned short* D = dst + (size_t)e * M * N;
    const int n0 = blockIdx.x * 32, m0 = blockIdx.y * 32;
    const int tx = threadIdx.x, ty = threadIdx.y;     // 32 x 8
#pragma unroll
    for (int i = 0; i < 32; i += 8)
        tile[ty + i][tx] = S[(size_t)(m0 + ty + i) * N + n0 + tx];
    __syncthreads();
#pragma unroll
    for (int i = 0; i < 32; i += 8)
        D[(size_t)(n0 + ty + i) * M + m0 + tx] = f2bf(tile[tx][ty + i]);
}

// ---------------------------------------------------------------- gating ----
__global__ void moe_gate_kernel(const float* __restrict__ x,
                                const float* __restrict__ gw,
                                int* __restrict__ counts,
                                int* __restrict__ tok_list,
                                float* __restrict__ gate_list) {
    int wid = threadIdx.x >> 5, lane = threadIdx.x & 31;
    int t = blockIdx.x * 8 + wid;
    if (t >= NT) return;
    float acc[NE];
#pragma unroll
    for (int e = 0; e < NE; ++e) acc[e] = 0.f;
    const float* xr = x + (size_t)t * ND;
    for (int d = lane; d < ND; d += 32) {
        float xv = xr[d];
        float4 g0 = *(const float4*)(gw + d * NE);
        float4 g1 = *(const float4*)(gw + d * NE + 4);
        acc[0] += xv * g0.x; acc[1] += xv * g0.y; acc[2] += xv * g0.z; acc[3] += xv * g0.w;
        acc[4] += xv * g1.x; acc[5] += xv * g1.y; acc[6] += xv * g1.z; acc[7] += xv * g1.w;
    }
#pragma unroll
    for (int e = 0; e < NE; ++e)
        for (int off = 16; off > 0; off >>= 1)
            acc[e] += __shfl_xor(acc[e], off, 32);
    if (lane == 0) {
        int b0 = 0;
        for (int e = 1; e < NE; ++e) if (acc[e] > acc[b0]) b0 = e;   // first-on-tie
        int b1 = -1;
        for (int e = 0; e < NE; ++e) {
            if (e == b0) continue;
            if (b1 < 0 || acc[e] > acc[b1]) b1 = e;
        }
        float e1 = expf(acc[b1] - acc[b0]);        // softmax over the top-2
        float inv = 1.f / (1.f + e1);
        int p0 = atomicAdd(&counts[b0], 1);
        tok_list[b0 * NT + p0] = t; gate_list[b0 * NT + p0] = inv;
        int p1 = atomicAdd(&counts[b1], 1);
        tok_list[b1 * NT + p1] = t; gate_list[b1 * NT + p1] = e1 * inv;
    }
}

__global__ void moe_scan_kernel(const int* __restrict__ counts, int* __restrict__ offsets) {
    if (threadIdx.x == 0 && blockIdx.x == 0) {
        int s = 0;
        for (int e = 0; e < NE; ++e) { offsets[e] = s; s += counts[e]; }
        offsets[NE] = s;
    }
}

// Tiling
#define BM 64
#define BN 128
#define BK 32
#define SA 36      // ffn1 A row stride: 72 B rows -> 8B-aligned ds_store_b64
#define SA2 40     // ffn2 A row stride: 80 B rows -> 16B-aligned async b128 dst
#define SBB 40     // B row stride (transposed bf16 weights): 80 B, async-aligned

// ---------------------------------------------------------------- FFN 1 -----
// H[offsets[e]+slot, f] = gelu( x[tok] @ W1[e] + b1[e] )   (bf16 out)
// B operand: async b128 copies from pre-transposed bf16 w1t[e]: [F][D].
__global__ void moe_ffn1_kernel(const float* __restrict__ x,
                                const unsigned short* __restrict__ w1t,
                                const float* __restrict__ b1,
                                const int* __restrict__ counts,
                                const int* __restrict__ offsets,
                                const int* __restrict__ tok_list,
                                unsigned short* __restrict__ H) {
    const int e  = blockIdx.z;
    const int Ne = counts[e];
    const int m0 = blockIdx.y * BM;
    if (m0 >= Ne) return;
    const int n0 = blockIdx.x * BN;

    __shared__ alignas(16) unsigned short As[2][BM * SA];
    __shared__ alignas(16) unsigned short Bs[2][BN * SBB];
    __shared__ int toks[BM];

    const int tid = threadIdx.x;
    const int lane = tid & 31, wid = tid >> 5;
    const int wm = wid & 1, wn = wid >> 1;           // 2x4 wave grid -> 64x128 tile
    const int half = lane >> 4, lr = lane & 15;

    if (tid < BM) {
        int slot = m0 + tid;
        toks[tid] = tok_list[e * NT + (slot < Ne ? slot : Ne - 1)];
    }
    __syncthreads();

    // A staging: 64x32 fp32 = 512 float4; p in {0,1}: i=tid+p*256, row=i>>3, seg=i&7
    const int ar0 = tid >> 3,         aseg0 = tid & 7;
    const int ar1 = (tid + 256) >> 3, aseg1 = (tid + 256) & 7;
    const float* abase0 = x + (size_t)toks[ar0] * ND + aseg0 * 4;
    const float* abase1 = x + (size_t)toks[ar1] * ND + aseg1 * 4;
    float4 ar[2];

    // B async: 128 rows x 64B = 512 x b128 chunks; p in {0,1}: i=tid+p*256
    const int bn0 = tid >> 2,          bc0 = tid & 3;
    const int bn1 = (tid + 256) >> 2,  bc1 = (tid + 256) & 3;
    const unsigned short* Wt = w1t + (size_t)e * NF * ND;
    const unsigned short* bg0 = Wt + (size_t)(n0 + bn0) * ND + bc0 * 8;
    const unsigned short* bg1 = Wt + (size_t)(n0 + bn1) * ND + bc1 * 8;
    const unsigned blds0[2] = { (unsigned)(size_t)&Bs[0][bn0 * SBB + bc0 * 8],
                                (unsigned)(size_t)&Bs[1][bn0 * SBB + bc0 * 8] };
    const unsigned blds1[2] = { (unsigned)(size_t)&Bs[0][bn1 * SBB + bc1 * 8],
                                (unsigned)(size_t)&Bs[1][bn1 * SBB + bc1 * 8] };

    v8f acc[2][2];
#pragma unroll
    for (int i = 0; i < 2; ++i)
#pragma unroll
        for (int j = 0; j < 2; ++j)
#pragma unroll
            for (int r = 0; r < 8; ++r) acc[i][j][r] = 0.f;

    auto issue_b = [&](int k0, int b) {
        async_b128(blds0[b], bg0 + k0);
        async_b128(blds1[b], bg1 + k0);
    };
    auto load_a = [&](int k0) {
        ar[0] = *(const float4*)(abase0 + k0);
        ar[1] = *(const float4*)(abase1 + k0);
    };
    auto store_a = [&](int b) {
        uint2 pa0 = make_uint2(cvtpk(ar[0].x, ar[0].y), cvtpk(ar[0].z, ar[0].w));
        uint2 pa1 = make_uint2(cvtpk(ar[1].x, ar[1].y), cvtpk(ar[1].z, ar[1].w));
        *(uint2*)&As[b][ar0 * SA + aseg0 * 4] = pa0;   // single ds_store_b64
        *(uint2*)&As[b][ar1 * SA + aseg1 * 4] = pa1;
    };
    auto compute = [&](int b) {
        union { v16bf v; unsigned int u[8]; } fa[2], fb[2];
#pragma unroll
        for (int mi = 0; mi < 2; ++mi) {
            int row = wm * 32 + mi * 16 + lr;
#pragma unroll
            for (int j = 0; j < 8; ++j)
                fa[mi].u[j] = *(const unsigned int*)&As[b][row * SA + kstart_of(j, half)];
        }
#pragma unroll
        for (int ni = 0; ni < 2; ++ni) {
            int col = wn * 32 + ni * 16 + lr;
#pragma unroll
            for (int j = 0; j < 8; ++j)
                fb[ni].u[j] = *(const unsigned int*)&Bs[b][col * SBB + kstart_of(j, half)];
        }
#pragma unroll
        for (int mi = 0; mi < 2; ++mi)
#pragma unroll
            for (int ni = 0; ni < 2; ++ni)
                acc[mi][ni] = __builtin_amdgcn_wmma_f32_16x16x32_bf16(
                    false, fa[mi].v, false, fb[ni].v, (short)0, acc[mi][ni], false, false);
    };

    const int NK = ND / BK;
    issue_b(0, 0);
    load_a(0);
    store_a(0);
    wait_async0();
    __syncthreads();
    for (int it = 0; it < NK; ++it) {
        if (it + 1 < NK) {
            issue_b((it + 1) * BK, (it + 1) & 1);     // async under WMMA
            load_a((it + 1) * BK);
        }
        compute(it & 1);
        if (it + 1 < NK) store_a((it + 1) & 1);
        wait_async0();
        __syncthreads();
    }

    const int goff = offsets[e];
#pragma unroll
    for (int mi = 0; mi < 2; ++mi)
#pragma unroll
        for (int ni = 0; ni < 2; ++ni)
#pragma unroll
            for (int r = 0; r < 8; ++r) {
                int row  = wm * 32 + mi * 16 + half * 8 + r;
                int slot = m0 + row;
                if (slot >= Ne) continue;
                int f = n0 + wn * 32 + ni * 16 + lr;
                float v = acc[mi][ni][r] + b1[e * NF + f];
                float g = 0.5f * v * (1.0f + erff(v * 0.70710678118654752f)); // exact GELU
                H[(size_t)(goff + slot) * NF + f] = f2bf(g);
            }
}

// ---------------------------------------------------------------- FFN 2 -----
// out[tok] += gate * ( H_row @ W2[e] + b2[e] )
// BOTH operands bf16 in memory -> fully async tile copies.
__global__ void moe_ffn2_kernel(const unsigned short* __restrict__ H,
                                const unsigned short* __restrict__ w2t,
                                const float* __restrict__ b2,
                                const int* __restrict__ counts,
                                const int* __restrict__ offsets,
                                const int* __restrict__ tok_list,
                                const float* __restrict__ gate_list,
                                float* __restrict__ out) {
    const int e  = blockIdx.z;
    const int Ne = counts[e];
    const int m0 = blockIdx.y * BM;
    if (m0 >= Ne) return;
    const int n0 = blockIdx.x * BN;

    __shared__ alignas(16) unsigned short As[2][BM * SA2];
    __shared__ alignas(16) unsigned short Bs[2][BN * SBB];

    const int tid = threadIdx.x;
    const int lane = tid & 31, wid = tid >> 5;
    const int wm = wid & 1, wn = wid >> 1;
    const int half = lane >> 4, lr = lane & 15;
    const int goff = offsets[e];

    // A: 64 rows x 32 bf16 = 64B/row; one b128 per thread.
    const int arow = tid >> 2, aseg = tid & 3;
    int aslot = m0 + arow; if (aslot >= Ne) aslot = Ne - 1;     // clamp, rows unused later
    const unsigned short* agbase = H + (size_t)(goff + aslot) * NF + aseg * 8;
    const unsigned alds[2] = { (unsigned)(size_t)&As[0][arow * SA2 + aseg * 8],
                               (unsigned)(size_t)&As[1][arow * SA2 + aseg * 8] };

    // B: 128 rows x 64B from w2t[e]: [D][F]; two b128 per thread.
    const int bn0 = tid >> 2,          bc0 = tid & 3;
    const int bn1 = (tid + 256) >> 2,  bc1 = (tid + 256) & 3;
    const unsigned short* Wt = w2t + (size_t)e * ND * NF;
    const unsigned short* bg0 = Wt + (size_t)(n0 + bn0) * NF + bc0 * 8;
    const unsigned short* bg1 = Wt + (size_t)(n0 + bn1) * NF + bc1 * 8;
    const unsigned blds0[2] = { (unsigned)(size_t)&Bs[0][bn0 * SBB + bc0 * 8],
                                (unsigned)(size_t)&Bs[1][bn0 * SBB + bc0 * 8] };
    const unsigned blds1[2] = { (unsigned)(size_t)&Bs[0][bn1 * SBB + bc1 * 8],
                                (unsigned)(size_t)&Bs[1][bn1 * SBB + bc1 * 8] };

    v8f acc[2][2];
#pragma unroll
    for (int i = 0; i < 2; ++i)
#pragma unroll
        for (int j = 0; j < 2; ++j)
#pragma unroll
            for (int r = 0; r < 8; ++r) acc[i][j][r] = 0.f;

    auto issue_tiles = [&](int k0, int b) {
        async_b128(alds[b],  agbase + k0);
        async_b128(blds0[b], bg0 + k0);
        async_b128(blds1[b], bg1 + k0);
    };
    auto compute = [&](int b) {
        union { v16bf v; unsigned int u[8]; } fa[2], fb[2];
#pragma unroll
        for (int mi = 0; mi < 2; ++mi) {
            int row = wm * 32 + mi * 16 + lr;
#pragma unroll
            for (int j = 0; j < 8; ++j)
                fa[mi].u[j] = *(const unsigned int*)&As[b][row * SA2 + kstart_of(j, half)];
        }
#pragma unroll
        for (int ni = 0; ni < 2; ++ni) {
            int col = wn * 32 + ni * 16 + lr;
#pragma unroll
            for (int j = 0; j < 8; ++j)
                fb[ni].u[j] = *(const unsigned int*)&Bs[b][col * SBB + kstart_of(j, half)];
        }
#pragma unroll
        for (int mi = 0; mi < 2; ++mi)
#pragma unroll
            for (int ni = 0; ni < 2; ++ni)
                acc[mi][ni] = __builtin_amdgcn_wmma_f32_16x16x32_bf16(
                    false, fa[mi].v, false, fb[ni].v, (short)0, acc[mi][ni], false, false);
    };

    const int NK = NF / BK;
    issue_tiles(0, 0);
    wait_async0();
    __syncthreads();
    for (int it = 0; it < NK; ++it) {
        if (it + 1 < NK) issue_tiles((it + 1) * BK, (it + 1) & 1);  // async under WMMA
        compute(it & 1);
        wait_async0();
        __syncthreads();
    }

    const int* tl = tok_list + e * NT;
    const float* gl = gate_list + e * NT;
#pragma unroll
    for (int mi = 0; mi < 2; ++mi)
#pragma unroll
        for (int ni = 0; ni < 2; ++ni)
#pragma unroll
            for (int r = 0; r < 8; ++r) {
                int row  = wm * 32 + mi * 16 + half * 8 + r;
                int slot = m0 + row;
                if (slot >= Ne) continue;
                int d = n0 + wn * 32 + ni * 16 + lr;
                float y = acc[mi][ni][r] + b2[e * ND + d];
                // exactly two commutative fp32 adds per out element -> deterministic
                atomicAdd(&out[(size_t)tl[slot] * ND + d], gl[slot] * y);
            }
}

// ---------------------------------------------------------------- launch ----
extern "C" void kernel_launch(void* const* d_in, const int* in_sizes, int n_in,
                              void* d_out, int out_size, void* d_ws, size_t ws_size,
                              hipStream_t stream) {
    const float* x    = (const float*)d_in[0];   // [T, D]
    const float* gw   = (const float*)d_in[1];   // [D, E]
    const float* w1   = (const float*)d_in[2];   // [E, D, F]
    const float* b1   = (const float*)d_in[3];   // [E, F]
    const float* w2   = (const float*)d_in[4];   // [E, F, D]
    const float* b2   = (const float*)d_in[5];   // [E, D]
    float* out = (float*)d_out;

    char* ws = (char*)d_ws;
    int*   counts    = (int*)ws;                           // 8 ints (+pad)
    int*   offsets   = (int*)(ws + 64);                    // 9 ints
    int*   tok_list  = (int*)(ws + 128);                   // E*T ints = 256 KB
    float* gate_list = (float*)(ws + 128 + (size_t)NE * NT * 4);
    unsigned short* H   = (unsigned short*)(ws + 524544);      // 16384 x 4096 bf16 = 134 MB
    unsigned short* w1t = (unsigned short*)(ws + 134742272);   // [E][F][D] bf16 = 67 MB
    unsigned short* w2t = (unsigned short*)(ws + 201851136);   // [E][D][F] bf16 = 67 MB

    hipMemsetAsync(counts, 0, 64, stream);
    hipMemsetAsync(out, 0, (size_t)out_size * sizeof(float), stream);

    // One-time weight transpose+convert (memory-bound, ~17us at 23.3 TB/s).
    moe_tcvt_kernel<<<dim3(NF / 32, ND / 32, NE), dim3(32, 8), 0, stream>>>(w1, w1t, ND, NF);
    moe_tcvt_kernel<<<dim3(ND / 32, NF / 32, NE), dim3(32, 8), 0, stream>>>(w2, w2t, NF, ND);

    moe_gate_kernel<<<NT / 8, 256, 0, stream>>>(x, gw, counts, tok_list, gate_list);
    moe_scan_kernel<<<1, 1, 0, stream>>>(counts, offsets);
    moe_ffn1_kernel<<<dim3(NF / BN, NT / BM, NE), 256, 0, stream>>>(
        x, w1t, b1, counts, offsets, tok_list, H);
    moe_ffn2_kernel<<<dim3(ND / BN, NT / BM, NE), 256, 0, stream>>>(
        H, w2t, b2, counts, offsets, tok_list, gate_list, out);
}
